// LigerFusedLinearCrossEntropy_62629213110396
// MI455X (gfx1250) — matrix operand order repaired and checked by reading, not compiled, not run on hardware
//
#include <hip/hip_runtime.h>
#include <hip/hip_bf16.h>

// ---------------------------------------------------------------------------
// Fused linear + cross-entropy, BT=8192, H=2048, V=32000 (fp32 in).
// Compute-bound at bf16 WMMA rate. 2x4 register blocking per wave:
//   per k-step: 2 A frags (global bf16) + 4 B frags (LDS) -> 8 wmmas.
// W chunk bf16 in LDS, K-block-major tiled layout so every B fragment is a
// ds_load_b128 with an immediate offset (one address add per k-step), padded
// 16B per 4 rows for conflict-free banking. x pre-converted to bf16 once.
// ---------------------------------------------------------------------------

#define BT      8192
#define HDIM    2048
#define VDIM    32000
#define VTILE   64            // W rows per workgroup (LDS-resident, bf16)
#define NWAVE   16
#define MTILE   512           // 16 waves x 32 rows per outer iteration
#define IGNORE_INDEX (-100)

// K-block-major LDS tiling (elements are __bf16):
//   block stride (32 K-values x 64 rows, padded): 2176 elems (4352 B)
//   row offset inside block: r*32 + (r/4)*8 elems  (16 B pad per 4 rows)
//   column-tile stride: 16 rows = 544 elems (1088 B, fits DS immediate)
#define KBLK_ELEMS   2176
#define TILE_ELEMS   544
#define LDS_W_ELEMS  ((HDIM / 32) * KBLK_ELEMS)   // 139264 elems = 278528 B

typedef __attribute__((ext_vector_type(16))) __bf16 v16bf;
typedef __attribute__((ext_vector_type(8)))  __bf16 v8bf;
typedef __attribute__((ext_vector_type(8)))  float  v8f;
typedef __attribute__((ext_vector_type(4)))  float  v4f;

union Frag16 { v16bf v; v8bf h[2]; };

#define WMMA_BF16(A, B, C) \
    __builtin_amdgcn_wmma_f32_16x16x32_bf16(false, (A), false, (B), (short)0, (C), false, false)

__global__ __launch_bounds__(256) void flce_zero_ws(float* ws, int n) {
    int i = blockIdx.x * 256 + threadIdx.x;
    if (i < n) ws[i] = 0.0f;
}

// x fp32 -> bf16, 8 elements per thread, exact grid coverage.
__global__ __launch_bounds__(256) void flce_cvt_x(const float* __restrict__ x,
                                                  __bf16* __restrict__ xb) {
    size_t i = (size_t)(blockIdx.x * 256 + threadIdx.x) * 8;
    v4f f0 = *(const v4f*)(x + i);
    v4f f1 = *(const v4f*)(x + i + 4);
    v8bf o;
    o[0] = (__bf16)f0.x; o[1] = (__bf16)f0.y; o[2] = (__bf16)f0.z; o[3] = (__bf16)f0.w;
    o[4] = (__bf16)f1.x; o[5] = (__bf16)f1.y; o[6] = (__bf16)f1.z; o[7] = (__bf16)f1.w;
    *(v8bf*)(xb + i) = o;
}

__device__ __forceinline__ void tile_epilogue(v8f acc, int rowbase, int half,
                                              int l16, int lab, int col,
                                              int rsbase, float* rs,
                                              float* __restrict__ tgt) {
    float e[8];
    #pragma unroll
    for (int j = 0; j < 8; ++j) {
        int labj = __shfl(lab, half * 8 + j, 16);                 // label broadcast
        if (labj == col) tgt[rowbase + half * 8 + j] = acc[j];    // unique writer
        e[j] = __expf(acc[j]);
    }
    #pragma unroll
    for (int m = 1; m < 16; m <<= 1) {
        #pragma unroll
        for (int j = 0; j < 8; ++j) e[j] += __shfl_xor(e[j], m, 32);
    }
    #pragma unroll
    for (int j = 0; j < 8; ++j) {
        if (l16 == j) atomicAdd(&rs[rsbase + half * 8 + j], e[j]);  // ds_add_f32
    }
}

__global__ __launch_bounds__(512) void flce_main(const __bf16* __restrict__ xb,
                                                 const int*    __restrict__ labels,
                                                 const float*  __restrict__ w,
                                                 float* __restrict__ rowsum,
                                                 float* __restrict__ tgt) {
    __shared__ __bf16 wlds[LDS_W_ELEMS];    // 278528 B: W chunk, bf16, K-block tiled
    __shared__ float  rs[MTILE];            // per-row partial sum-exp within WG

    const int vbase = blockIdx.x * VTILE;
    const int tid   = threadIdx.x;
    const int wv    = tid >> 5;             // 0..15
    const int lane  = tid & 31;
    const int half  = lane >> 4;            // 0: lanes 0-15, 1: lanes 16-31
    const int l16   = lane & 15;

    // ---- Phase 1: stage W chunk fp32 -> bf16 into tiled LDS, once ----
    for (int r = wv; r < VTILE; r += NWAVE) {
        const float* src  = w + (size_t)(vbase + r) * HDIM;
        const int rowoff  = r * 32 + (r >> 2) * 8;     // elems within K-block
        for (int c = lane * 4; c < HDIM; c += 32 * 4) {
            v4f f = *(const v4f*)(src + c);
            __bf16* dst = wlds + (c >> 5) * KBLK_ELEMS + rowoff + (c & 31);
            dst[0] = (__bf16)f.x;
            dst[1] = (__bf16)f.y;
            dst[2] = (__bf16)f.z;
            dst[3] = (__bf16)f.w;
        }
    }
    __syncthreads();

    // Per-lane B base: row = l16 (within tile), K offset half*8
    const __bf16* bbase0 = wlds + l16 * 32 + (l16 >> 2) * 8 + half * 8;

    // ---- Phase 2: stream all x rows (bf16, L2-resident), 2x4 blocking ----
    for (int m0 = 0; m0 < BT; m0 += MTILE) {
        rs[tid] = 0.0f;
        __syncthreads();

        const int row0 = m0 + wv * 32 + l16;           // strip 0 A row
        const int lab0 = labels[row0];
        const int lab1 = labels[row0 + 16];            // strip 1
        const __bf16* xrow0 = xb + (size_t)row0 * HDIM + half * 8;
        const __bf16* xrow1 = xrow0 + (size_t)16 * HDIM;

        v8f acc[2][4];
        #pragma unroll
        for (int a = 0; a < 2; ++a)
            #pragma unroll
            for (int vt = 0; vt < 4; ++vt) acc[a][vt] = (v8f){};

        const __bf16* bb = bbase0;
        #pragma unroll 2
        for (int k0 = 0; k0 < HDIM; k0 += 32) {
            Frag16 A0, A1;
            A0.h[0] = *(const v8bf*)(xrow0 + k0);
            A0.h[1] = *(const v8bf*)(xrow0 + k0 + 16);
            A1.h[0] = *(const v8bf*)(xrow1 + k0);
            A1.h[1] = *(const v8bf*)(xrow1 + k0 + 16);

            #pragma unroll
            for (int vt = 0; vt < 4; ++vt) {
                Frag16 B;   // immediate DS offsets: vt*1088 + {0,32} bytes
                B.h[0] = *(const v8bf*)(bb + vt * TILE_ELEMS);
                B.h[1] = *(const v8bf*)(bb + vt * TILE_ELEMS + 16);
                acc[0][vt] = WMMA_BF16(A0.v, B.v, acc[0][vt]);
                acc[1][vt] = WMMA_BF16(A1.v, B.v, acc[1][vt]);
            }
            bb += KBLK_ELEMS;                          // one address add per k-step
        }

        #pragma unroll
        for (int a = 0; a < 2; ++a) {
            const int rowbase = m0 + wv * 32 + a * 16;
            const int rsbase  = wv * 32 + a * 16;
            const int lab     = a ? lab1 : lab0;
            #pragma unroll
            for (int vt = 0; vt < 4; ++vt) {
                tile_epilogue(acc[a][vt], rowbase, half, l16, lab,
                              vbase + vt * 16 + l16, rsbase, rs, tgt);
            }
        }
        __syncthreads();

        atomicAdd(&rowsum[m0 + tid], rs[tid]);
        __syncthreads();
    }
}

__global__ __launch_bounds__(256) void flce_reduce(const float* __restrict__ rowsum,
                                                   const float* __restrict__ tgt,
                                                   const int*   __restrict__ labels,
                                                   float* __restrict__ partial) {
    __shared__ float s_nll[256];
    __shared__ float s_cnt[256];
    int i = blockIdx.x * 256 + threadIdx.x;
    float nll = 0.0f, cnt = 0.0f;
    if (i < BT) {
        int lab = labels[i];
        if (lab != IGNORE_INDEX) {
            nll = __logf(rowsum[i]) - tgt[i];
            cnt = 1.0f;
        }
    }
    s_nll[threadIdx.x] = nll;
    s_cnt[threadIdx.x] = cnt;
    __syncthreads();
    for (int off = 128; off > 0; off >>= 1) {
        if (threadIdx.x < off) {
            s_nll[threadIdx.x] += s_nll[threadIdx.x + off];
            s_cnt[threadIdx.x] += s_cnt[threadIdx.x + off];
        }
        __syncthreads();
    }
    if (threadIdx.x == 0) {
        atomicAdd(&partial[0], s_nll[0]);
        atomicAdd(&partial[1], s_cnt[0]);
    }
}

__global__ void flce_final(const float* __restrict__ partial, float* __restrict__ out) {
    out[0] = partial[0] / fmaxf(partial[1], 1.0f);
}

extern "C" void kernel_launch(void* const* d_in, const int* in_sizes, int n_in,
                              void* d_out, int out_size, void* d_ws, size_t ws_size,
                              hipStream_t stream) {
    const float* x      = (const float*)d_in[0];   // hidden_states [BT, H]
    const int*   labels = (const int*)d_in[1];     // labels [BT]
    const float* w      = (const float*)d_in[2];   // weight [V, H]
    float* out = (float*)d_out;

    // workspace layout
    float*  rowsum  = (float*)d_ws;                              // [BT]
    float*  tgt     = rowsum + BT;                               // [BT]
    float*  partial = tgt + BT;                                  // [2]
    __bf16* xb      = (__bf16*)((char*)d_ws + (128 << 10));      // [BT*H] bf16 (~33.5 MB)

    const int nzero = 2 * BT + 2;
    flce_zero_ws<<<(nzero + 255) / 256, 256, 0, stream>>>(rowsum, nzero);
    flce_cvt_x<<<(BT * HDIM) / (256 * 8), 256, 0, stream>>>(x, xb);
    flce_main<<<VDIM / VTILE, 512, 0, stream>>>(xb, labels, w, rowsum, tgt);
    flce_reduce<<<BT / 256, 256, 0, stream>>>(rowsum, tgt, labels, partial);
    flce_final<<<1, 1, 0, stream>>>(partial, out);
}
